// GNNLSTM_21337397527215
// MI455X (gfx1250) — compile-verified, matrix operand order checked by reading, not run on hardware
//
#include <hip/hip_runtime.h>

#define DEVI static __device__ __forceinline__

typedef __attribute__((ext_vector_type(16))) __bf16 bf16x16;
typedef __attribute__((ext_vector_type(8)))  float  v8f;

namespace {

// ---------------------------------------------------------------------------
// bf16 helpers / WMMA fragment plumbing (wave32, v_wmma_f32_16x16x32_bf16)
// ---------------------------------------------------------------------------
union FragBF {
  bf16x16 v;
  uint4 q[2];
  unsigned short h[16];
};

DEVI unsigned short f2bf(float f) {
  union { float f; unsigned u; } a; a.f = f;
  unsigned u = a.u + 0x7FFFu + ((a.u >> 16) & 1u);   // round-to-nearest-even
  return (unsigned short)(u >> 16);
}

// 16x32 bf16 fragment: lane l holds row l%16; lanes 0-15 K={0..7,16..23},
// lanes 16-31 K={8..15,24..31}: two contiguous 16B chunks per lane.
// Works for global or LDS `base` (address-space inferred after inlining).
DEVI void frag_ld_bf(FragBF& fr, const unsigned short* base, int row0, int ld,
                     int k0, int lane) {
  int r = lane & 15, half = lane >> 4;
  const unsigned short* p = base + (size_t)(row0 + r) * ld + (k0 + half * 8);
  fr.q[0] = *(const uint4*)(p);
  fr.q[1] = *(const uint4*)(p + 16);
}

DEVI v8f wmma_bf16(const FragBF& a, const FragBF& b, v8f c) {
  return __builtin_amdgcn_wmma_f32_16x16x32_bf16(
      false, a.v, false, b.v, (short)0, c, false, false);
}

DEVI float sigm(float x) { return 1.0f / (1.0f + __expf(-x)); }

// ---------------------------------------------------------------------------
// f32 -> bf16 bulk conversion (vectorized, one pass over the weights)
// ---------------------------------------------------------------------------
__global__ void gnl_cvt_bf4(const float* __restrict__ src,
                            unsigned short* __restrict__ dst, int n4) {
  int i = blockIdx.x * blockDim.x + threadIdx.x;
  if (i < n4) {
    float4 x = ((const float4*)src)[i];
    union { unsigned short h[4]; uint2 u; } o;
    o.h[0] = f2bf(x.x); o.h[1] = f2bf(x.y);
    o.h[2] = f2bf(x.z); o.h[3] = f2bf(x.w);
    ((uint2*)dst)[i] = o.u;
  }
}

// ---------------------------------------------------------------------------
// Graph preprocessing: degree, normalized dense adjacency (with self loops)
// ---------------------------------------------------------------------------
__global__ void gnl_deg(const int* ei, float* deg) {
  int e = blockIdx.x * blockDim.x + threadIdx.x;
  if (e < 1024) atomicAdd(&deg[ei[1024 + e]], 1.0f);   // dst row
}

__global__ void gnl_adj_edges(const int* ei, const float* deg, float* A) {
  int e = blockIdx.x * blockDim.x + threadIdx.x;
  if (e < 1024) {
    int s = ei[e], d = ei[1024 + e];
    float nv = rsqrtf(deg[s] + 1.0f) * rsqrtf(deg[d] + 1.0f);
    atomicAdd(&A[d * 256 + s], nv);
  }
}

__global__ void gnl_adj_fin(const float* A, const float* deg, unsigned short* Abf) {
  int idx = blockIdx.x * blockDim.x + threadIdx.x;   // 65536
  int n = idx >> 8, m = idx & 255;
  float v = A[idx];
  if (n == m) { float di = rsqrtf(deg[n] + 1.0f); v += di * di; }
  Abf[idx] = f2bf(v);
}

__global__ void gnl_wg2t(const float* Wg2, unsigned short* WT) {
  int idx = blockIdx.x * blockDim.x + threadIdx.x;   // 4096: WT[n*64+k]=Wg2[k*64+n]
  int n = idx >> 6, k = idx & 63;
  WT[idx] = f2bf(Wg2[k * 64 + n]);
}

// embW[n,c] = relu(node_feat @ W_embed + b_embed)[n,:] @ W_g1[8:,c]
__global__ void gnl_embw(const float* nf, const float* We, const float* be,
                         const float* Wg1, float* embW) {
  __shared__ float se[64];
  int n = blockIdx.x, c = threadIdx.x;
  float a = be[c];
  for (int s = 0; s < 16; s++) a += nf[n * 16 + s] * We[s * 64 + c];
  se[c] = fmaxf(a, 0.0f);
  __syncthreads();
  float v = 0.0f;
  for (int h = 0; h < 64; h++) v += se[h] * Wg1[(8 + h) * 64 + c];
  embW[n * 64 + c] = v;
}

// ---------------------------------------------------------------------------
// Fully fused GCN (both layers) for one (b,l) slice per block, LDS-staged.
//   A: Hraw^T in LDS  ->  B: H1 = relu(Aadj@Hraw+b1) in LDS
//   C: H2raw^T = H1@Wg2 in LDS -> D: seq = relu(Aadj@H2raw+b2) to global
// 256 threads (8 waves); LDS strides padded for conflict-free ds_load_b128.
// ---------------------------------------------------------------------------
#define SB_LD 264   // [c][m] stride (ushorts): 528B, 16B-aligned, bank step 4
#define SH_LD 72    // [m][c] stride (ushorts): 144B, 16B-aligned, bank step 36

__global__ void gnl_gcn_fused(const float* __restrict__ x_seq,
                              const float* __restrict__ Wg1,
                              const float* __restrict__ embW,
                              const unsigned short* __restrict__ Abf,
                              const unsigned short* __restrict__ WG2T,
                              const float* __restrict__ b1,
                              const float* __restrict__ b2,
                              unsigned short* __restrict__ seqbf) {
  __shared__ __align__(16) unsigned short sB[64 * SB_LD];    // HrawT / H2rawT
  __shared__ __align__(16) unsigned short sH1[256 * SH_LD];  // H1 row-major
  int s = blockIdx.x;
  int tid = threadIdx.x;
  int wave = tid >> 5, lane = tid & 31;

  // ---- Stage A: HrawT[c][m] = x_seq[s,m,:8]@Wg1[:8,c] + embW[m,c]
  {
    int m = tid;
    const float* xp = x_seq + ((size_t)s * 256 + m) * 8;
    float xv[8];
#pragma unroll
    for (int f = 0; f < 8; f++) xv[f] = xp[f];
    for (int c = 0; c < 64; c++) {
      float v = embW[m * 64 + c];
#pragma unroll
      for (int f = 0; f < 8; f++) v += xv[f] * Wg1[f * 64 + c];
      sB[c * SB_LD + m] = f2bf(v);
    }
  }
  __syncthreads();

  // ---- Stage B: H1 = relu(Aadj @ Hraw + b1)   (M=256,N=64,K=256)
  for (int i = 0; i < 8; i++) {
    int tile = wave * 8 + i;
    int mt = tile >> 2, nt = tile & 3;
    v8f acc = {};
    for (int k0 = 0; k0 < 256; k0 += 32) {
      FragBF a, b;
      frag_ld_bf(a, Abf, mt * 16, 256, k0, lane);
      frag_ld_bf(b, sB, nt * 16, SB_LD, k0, lane);
      acc = wmma_bf16(a, b, acc);
    }
    int m0 = mt * 16 + 8 * (lane >> 4), n = nt * 16 + (lane & 15);
    float bias = b1[n];
#pragma unroll
    for (int r = 0; r < 8; r++)
      sH1[(m0 + r) * SH_LD + n] = f2bf(fmaxf(acc[r] + bias, 0.0f));
  }
  __syncthreads();

  // ---- Stage C: H2rawT[c'][m] = (H1 @ Wg2)[m,c']   (K=64), overwrite sB
  for (int i = 0; i < 8; i++) {
    int tile = wave * 8 + i;
    int mt = tile >> 2, nt = tile & 3;
    v8f acc = {};
    for (int k0 = 0; k0 < 64; k0 += 32) {
      FragBF a, b;
      frag_ld_bf(a, sH1, mt * 16, SH_LD, k0, lane);
      frag_ld_bf(b, WG2T, nt * 16, 64, k0, lane);
      acc = wmma_bf16(a, b, acc);
    }
    int m0 = mt * 16 + 8 * (lane >> 4), cp = nt * 16 + (lane & 15);
#pragma unroll
    for (int r = 0; r < 8; r++)
      sB[cp * SB_LD + (m0 + r)] = f2bf(acc[r]);
  }
  __syncthreads();

  // ---- Stage D: seq row = relu(Aadj @ H2raw + b2), LSTM-ready layout
  int bidx = s / 96, l = s - bidx * 96;                 // s = b*L + l
  unsigned short* outr = seqbf + (size_t)(l * 16 + bidx) * 16384;
  for (int i = 0; i < 8; i++) {
    int tile = wave * 8 + i;
    int mt = tile >> 2, nt = tile & 3;
    v8f acc = {};
    for (int k0 = 0; k0 < 256; k0 += 32) {
      FragBF a, b;
      frag_ld_bf(a, Abf, mt * 16, 256, k0, lane);
      frag_ld_bf(b, sB, nt * 16, SB_LD, k0, lane);
      acc = wmma_bf16(a, b, acc);
    }
    int m0 = mt * 16 + 8 * (lane >> 4), n = nt * 16 + (lane & 15);
    float bias = b2[n];
#pragma unroll
    for (int r = 0; r < 8; r++) {
      float v = fmaxf(acc[r] + bias, 0.0f);
      outr[(size_t)(m0 + r) * 64 + n] = f2bf(v);        // col = node*64 + channel
    }
  }
}

// ---------------------------------------------------------------------------
// Pure-bf16 GEMM: C(MxN,f32) = A(bf16 MxK) @ Bt(bf16 NxK)^T + bias0 + bias1
// grid = (M/16)*(N/64) x 128 threads; 4 waves share A rows (L1 reuse).
// ---------------------------------------------------------------------------
__global__ void gnl_gemm_abb(const unsigned short* __restrict__ A, int lda,
                             const unsigned short* __restrict__ Bt, int ldb,
                             const float* __restrict__ bias0,
                             const float* __restrict__ bias1,
                             float* __restrict__ C, int ldc, int ntq_count, int K) {
  int bid = blockIdx.x;
  int mt = bid / ntq_count, ntq = bid - mt * ntq_count;
  int wave = threadIdx.x >> 5, lane = threadIdx.x & 31;
  int nt = ntq * 4 + wave;
  v8f acc = {};
  for (int k0 = 0; k0 < K; k0 += 32) {
    FragBF a, b;
    frag_ld_bf(a, A, mt * 16, lda, k0, lane);
    frag_ld_bf(b, Bt, nt * 16, ldb, k0, lane);
    acc = wmma_bf16(a, b, acc);
  }
  int m0 = mt * 16 + 8 * (lane >> 4);
  int n = nt * 16 + (lane & 15);
  float bb = bias0[n] + bias1[n];
#pragma unroll
  for (int r = 0; r < 8; r++)
    C[(size_t)(m0 + r) * ldc + n] = acc[r] + bb;
}

// ---------------------------------------------------------------------------
// Persistent LSTM scan: one launch runs all 96 steps. 32 blocks x 128 threads
// (block wg owns hidden cols [wg*16,wg*16+16), wave w = gate w in i,f,g,o).
// Cell state c lives in registers; h ping-pongs between two global bf16 bufs;
// cross-block step barrier = release atomicAdd + acquire spin (s_sleep).
// ---------------------------------------------------------------------------
__global__ void gnl_lstm_scan(const float* __restrict__ G,       // 96*16 x 2048
                              const unsigned short* __restrict__ Whh_bf, // 2048x512
                              unsigned short* hbufA, unsigned short* hbufB,
                              unsigned short* __restrict__ hs,   // 96*16 x 512
                              float* hlast,                      // 16x512 or null
                              int* sync) {
  __shared__ float gsm[4][16][16];
  int tid = threadIdx.x;
  int wave = tid >> 5, lane = tid & 31;
  int jb = blockIdx.x * 16;
  int nbase = wave * 512 + jb;
  float creg0 = 0.0f, creg1 = 0.0f;

  for (int t = 0; t < 96; t++) {
    const unsigned short* hp = (t & 1) ? hbufB : hbufA;
    unsigned short* hn = (t & 1) ? hbufA : hbufB;

    v8f acc = {};
    for (int k0 = 0; k0 < 512; k0 += 32) {
      FragBF a, b;
      frag_ld_bf(a, hp, 0, 512, k0, lane);              // 16 batch rows
      frag_ld_bf(b, Whh_bf, nbase, 512, k0, lane);
      acc = wmma_bf16(a, b, acc);
    }
    const float* Grow = G + (size_t)t * 16 * 2048;
    int b0 = 8 * (lane >> 4), col = lane & 15, n = nbase + col;
#pragma unroll
    for (int r = 0; r < 8; r++)
      gsm[wave][b0 + r][col] = acc[r] + Grow[(size_t)(b0 + r) * 2048 + n];
    __syncthreads();

#pragma unroll
    for (int u = 0; u < 2; u++) {
      int e = tid + u * 128;                            // e = batch*16 + j
      int bb = e >> 4, j = e & 15;
      float gi = gsm[0][bb][j], gf = gsm[1][bb][j];
      float gg = gsm[2][bb][j], go = gsm[3][bb][j];
      float cprev = u ? creg1 : creg0;
      float cv = sigm(gf) * cprev + sigm(gi) * tanhf(gg);
      if (u) creg1 = cv; else creg0 = cv;
      float hv = sigm(go) * tanhf(cv);
      unsigned short hb = f2bf(hv);
      int idx = bb * 512 + jb + j;
      hn[idx] = hb;
      hs[(size_t)t * 8192 + idx] = hb;
      if (hlast && t == 95) hlast[idx] = hv;
    }

    // ---- device-scope step barrier ----
    __threadfence();
    __syncthreads();
    if (tid == 0)
      __hip_atomic_fetch_add(sync, 1, __ATOMIC_RELEASE, __HIP_MEMORY_SCOPE_AGENT);
    while (__hip_atomic_load(sync, __ATOMIC_ACQUIRE, __HIP_MEMORY_SCOPE_AGENT)
           < 32 * (t + 1))
      __builtin_amdgcn_s_sleep(2);
    __syncthreads();
  }
}

// ---------------------------------------------------------------------------
// Head (tiny, f32 VALU): tmp = relu(last@Wh1+bh1); out = tmp@Wh2+bh2
// ---------------------------------------------------------------------------
__global__ void gnl_head1(const float* hlast, const float* Wh1, const float* bh1,
                          float* tmp) {
  int t = blockIdx.x * blockDim.x + threadIdx.x;        // 8192
  int b = t >> 9, j = t & 511;
  float v = bh1[j];
  for (int k = 0; k < 512; k++) v += hlast[b * 512 + k] * Wh1[k * 512 + j];
  tmp[t] = fmaxf(v, 0.0f);
}

__global__ void gnl_head2(const float* tmp, const float* Wh2, const float* bh2,
                          float* out) {
  int t = blockIdx.x * blockDim.x + threadIdx.x;        // 196608
  int b = t / 12288, m = t - b * 12288;
  float v = bh2[m];
  for (int k = 0; k < 512; k++) v += tmp[b * 512 + k] * Wh2[(size_t)k * 12288 + m];
  out[t] = v;
}

}  // namespace

extern "C" void kernel_launch(void* const* d_in, const int* in_sizes, int n_in,
                              void* d_out, int out_size, void* d_ws, size_t ws_size,
                              hipStream_t stream) {
  (void)in_sizes; (void)n_in; (void)out_size; (void)ws_size;
  const float* x_seq     = (const float*)d_in[0];
  const float* node_feat = (const float*)d_in[2];
  const int*   edge_index= (const int*)d_in[3];
  const float* W_embed   = (const float*)d_in[4];
  const float* b_embed   = (const float*)d_in[5];
  const float* W_g1      = (const float*)d_in[6];
  const float* b_g1      = (const float*)d_in[7];
  const float* W_g2      = (const float*)d_in[8];
  const float* b_g2      = (const float*)d_in[9];
  const float* Wih0      = (const float*)d_in[10];
  const float* Whh0      = (const float*)d_in[11];
  const float* bih0      = (const float*)d_in[12];
  const float* bhh0      = (const float*)d_in[13];
  const float* Wih1      = (const float*)d_in[14];
  const float* Whh1      = (const float*)d_in[15];
  const float* bih1      = (const float*)d_in[16];
  const float* bhh1      = (const float*)d_in[17];
  const float* Wh1       = (const float*)d_in[18];
  const float* bh1       = (const float*)d_in[19];
  const float* Wh2       = (const float*)d_in[20];
  const float* bh2       = (const float*)d_in[21];
  float* out = (float*)d_out;

  char* w = (char*)d_ws;
  size_t off = 0;
  auto take = [&](size_t bytes) -> void* {
    void* p = w + off;
    off = (off + bytes + 255) & ~(size_t)255;
    return p;
  };
  float*          deg   = (float*)take(256 * 4);
  float*          Aadj  = (float*)take(65536 * 4);
  unsigned short* Abf   = (unsigned short*)take(65536 * 2);
  float*          embW  = (float*)take(256 * 64 * 4);
  unsigned short* WG2T  = (unsigned short*)take(64 * 64 * 2);
  unsigned short* seqbf = (unsigned short*)take((size_t)1536 * 16384 * 2);
  unsigned short* Wih0b = (unsigned short*)take((size_t)2048 * 16384 * 2);
  unsigned short* Wih1b = (unsigned short*)take((size_t)2048 * 512 * 2);
  unsigned short* Whh0b = (unsigned short*)take((size_t)2048 * 512 * 2);
  unsigned short* Whh1b = (unsigned short*)take((size_t)2048 * 512 * 2);
  float*          G0    = (float*)take((size_t)1536 * 2048 * 4);
  float*          G1    = (float*)take((size_t)1536 * 2048 * 4);
  unsigned short* hs1   = (unsigned short*)take(1536 * 512 * 2);
  unsigned short* hs2   = (unsigned short*)take(1536 * 512 * 2);
  unsigned short* h1A   = (unsigned short*)take(16 * 512 * 2);
  unsigned short* h1B   = (unsigned short*)take(16 * 512 * 2);
  unsigned short* h2A   = (unsigned short*)take(16 * 512 * 2);
  unsigned short* h2B   = (unsigned short*)take(16 * 512 * 2);
  float*          hlast = (float*)take(16 * 512 * 4);
  float*          tmpv  = (float*)take(16 * 512 * 4);
  int*            sync1 = (int*)take(256);
  int*            sync2 = (int*)take(256);

  hipMemsetAsync(deg, 0, 256 * 4, stream);
  hipMemsetAsync(Aadj, 0, 65536 * 4, stream);
  hipMemsetAsync(h1A, 0, 16 * 512 * 2, stream);
  hipMemsetAsync(h2A, 0, 16 * 512 * 2, stream);
  hipMemsetAsync(sync1, 0, 256, stream);
  hipMemsetAsync(sync2, 0, 256, stream);

  // Weight conversions (one pass each, bandwidth-bound)
  gnl_cvt_bf4<<<(2048 * 16384 / 4 + 255) / 256, 256, 0, stream>>>(Wih0, Wih0b,
                                                                  2048 * 16384 / 4);
  gnl_cvt_bf4<<<(2048 * 512 / 4 + 255) / 256, 256, 0, stream>>>(Wih1, Wih1b,
                                                                2048 * 512 / 4);
  gnl_cvt_bf4<<<(2048 * 512 / 4 + 255) / 256, 256, 0, stream>>>(Whh0, Whh0b,
                                                                2048 * 512 / 4);
  gnl_cvt_bf4<<<(2048 * 512 / 4 + 255) / 256, 256, 0, stream>>>(Whh1, Whh1b,
                                                                2048 * 512 / 4);

  // Graph preprocessing
  gnl_deg<<<4, 256, 0, stream>>>(edge_index, deg);
  gnl_adj_edges<<<4, 256, 0, stream>>>(edge_index, deg, Aadj);
  gnl_adj_fin<<<256, 256, 0, stream>>>(Aadj, deg, Abf);
  gnl_wg2t<<<16, 256, 0, stream>>>(W_g2, WG2T);
  gnl_embw<<<256, 64, 0, stream>>>(node_feat, W_embed, b_embed, W_g1, embW);

  // Fused 2-layer GCN, one slice per block
  gnl_gcn_fused<<<1536, 256, 0, stream>>>(x_seq, W_g1, embW, Abf, WG2T,
                                          b_g1, b_g2, seqbf);

  // Dominant GEMM: G0 = seq(1536x16384) @ Wih0^T + (bih0+bhh0)
  gnl_gemm_abb<<<96 * 32, 128, 0, stream>>>(seqbf, 16384, Wih0b, 16384,
                                            bih0, bhh0, G0, 2048, 32, 16384);

  // Layer-1 recurrence: single persistent scan
  gnl_lstm_scan<<<32, 128, 0, stream>>>(G0, Whh0b, h1A, h1B, hs1,
                                        (float*)nullptr, sync1);

  // G1 = hs1(1536x512) @ Wih1^T + (bih1+bhh1)
  gnl_gemm_abb<<<96 * 32, 128, 0, stream>>>(hs1, 512, Wih1b, 512,
                                            bih1, bhh1, G1, 2048, 32, 512);

  // Layer-2 recurrence
  gnl_lstm_scan<<<32, 128, 0, stream>>>(G1, Whh1b, h2A, h2B, hs2, hlast, sync2);

  // Head
  gnl_head1<<<32, 256, 0, stream>>>(hlast, Wh1, bh1, tmpv);
  gnl_head2<<<768, 256, 0, stream>>>(tmpv, Wh2, bh2, out);
}
